// TransformerBlock_25280177504279
// MI455X (gfx1250) — matrix operand order repaired and checked by reading, not compiled, not run on hardware
//
#include <hip/hip_runtime.h>
#include <math.h>

// ---------------------------------------------------------------------------
// Transformer block for MI455X (gfx1250, wave32, WMMA bf16, async-LDS staging).
// ---------------------------------------------------------------------------

#define BB   2
#define LL   4096
#define DE   512
#define NH   8
#define HD   64
#define DFF  2048
#define MTOT (BB * LL)

typedef __attribute__((ext_vector_type(16))) __bf16 v16bf;
typedef __attribute__((ext_vector_type(8)))  float  v8f;
typedef int v4i_g __attribute__((vector_size(16)));   // GCC-style, matches builtin

#if defined(__has_builtin)
#  if __has_builtin(__builtin_amdgcn_global_load_async_to_lds_b128)
#    define HAVE_ASYNC_LDS 1
#  endif
#endif
#ifndef HAVE_ASYNC_LDS
#  define HAVE_ASYNC_LDS 0
#endif

union Frag {
    v16bf v;
    uint4 u[2];
};

__device__ __forceinline__ unsigned short f2bf(float f) {
    union { float f; unsigned u; } x; x.f = f;
    unsigned r = x.u + 0x7FFFu + ((x.u >> 16) & 1u);   // round-to-nearest-even
    return (unsigned short)(r >> 16);
}

__device__ __forceinline__ v8f wmma_bf16(const Frag& a, const Frag& b, v8f c) {
    return __builtin_amdgcn_wmma_f32_16x16x32_bf16(
        /*neg_a=*/false, a.v, /*neg_b=*/false, b.v,
        /*c_mod=*/(short)0, c, /*reuse_a=*/false, /*reuse_b=*/false);
}

// 16-byte global -> LDS copy: async (ASYNCcnt) when the builtin exists.
__device__ __forceinline__ void copy16_g2l(const unsigned short* gp, unsigned short* lp) {
#if HAVE_ASYNC_LDS
    __builtin_amdgcn_global_load_async_to_lds_b128(
        (__attribute__((address_space(1))) v4i_g*)((const __attribute__((address_space(1))) void*)gp),
        (__attribute__((address_space(3))) v4i_g*)((__attribute__((address_space(3))) void*)lp),
        0, 0);
#else
    *(uint4*)lp = *(const uint4*)gp;
#endif
}

// ---------------------------------------------------------------------------
// f32 -> bf16 convert (grid-stride)
// ---------------------------------------------------------------------------
__global__ void cvt_f32_bf16(const float* __restrict__ in,
                             unsigned short* __restrict__ out, int n) {
    int i = blockIdx.x * blockDim.x + threadIdx.x;
    int stride = gridDim.x * blockDim.x;
    for (; i < n; i += stride) out[i] = f2bf(in[i]);
}

// ---------------------------------------------------------------------------
// WMMA GEMM:  C[M,N] = A[M,K] (bf16 row-major) @ W[K,N] (bf16) + bias
// Block: 256 threads = 8 waves (4 M-waves x 2 N-waves); wave tile 32x64.
// Block tile 128(M) x 128(N).  B K-slab (32x128) staged in LDS, double-
// buffered via GLOBAL_LOAD_ASYNC_TO_LDS_B128 (ASYNCcnt) when available.
// emode: 0 = f32 out [M,N]
//        1 = exact GELU -> bf16 out [M,N]
//        2 = head-split bf16 out [B,H,L,64]   (Q / V)
//        3 = head-split transposed bf16 out [B,H,64,L]  (K^T)
// ---------------------------------------------------------------------------
#define BS_STRIDE 136   // 128 cols + 8 pad halfs: 16B-aligned rows, bank-swept

__global__ __launch_bounds__(256) void gemm_wmma_bf16(
    const unsigned short* __restrict__ A,
    const unsigned short* __restrict__ W,
    const float* __restrict__ bias,
    int M, int N, int K, int emode,
    float* __restrict__ outf,
    unsigned short* __restrict__ outh)
{
    __shared__ unsigned short Bs[2][32 * BS_STRIDE];

    const int lane = threadIdx.x & 31;
    const int wid  = threadIdx.x >> 5;
    const int wm   = wid >> 1;                 // 0..3
    const int wn   = wid & 1;                  // 0..1
    const int bm   = blockIdx.x * 128 + wm * 32;
    const int bn0  = blockIdx.y * 128;         // block N base
    const int wbn  = bn0 + wn * 64;            // wave N base
    const int g    = lane >> 4;
    const int ml   = lane & 15;

    // Staging: 32 K-rows x 128 cols = 512 x 16B chunks; 2 chunks per thread.
    const int cid0 = threadIdx.x * 2;

    v8f acc[2][4];
    v8f zero = {};
    #pragma unroll
    for (int mt = 0; mt < 2; ++mt)
        #pragma unroll
        for (int nt = 0; nt < 4; ++nt) acc[mt][nt] = zero;

    auto issue_stage = [&](int k0, int buf) {
        #pragma unroll
        for (int j = 0; j < 2; ++j) {
            const int cid = cid0 + j;
            const int kr  = cid >> 4;          // 16 chunks per row
            const int cc  = (cid & 15) * 8;    // half offset within row
            const unsigned short* gp = W + (size_t)(k0 + kr) * N + bn0 + cc;
            unsigned short* lp = &Bs[buf][kr * BS_STRIDE + cc];
            copy16_g2l(gp, lp);
        }
    };

    int buf = 0;
    issue_stage(0, 0);

    for (int k0 = 0; k0 < K; k0 += 32) {
        const bool have_next = (k0 + 32) < K;
        if (have_next) issue_stage(k0 + 32, buf ^ 1);
#if HAVE_ASYNC_LDS
        if (have_next) asm volatile("s_wait_asynccnt 2" ::: "memory");
        else           asm volatile("s_wait_asynccnt 0" ::: "memory");
#endif
        __syncthreads();   // current-stage B slab visible to all waves

        // A fragments: direct pipelined global loads (2x16B chunks per lane).
        Frag af[2];
        #pragma unroll
        for (int mt = 0; mt < 2; ++mt) {
            const unsigned short* p = A + (size_t)(bm + mt * 16 + ml) * K + k0 + g * 8;
            af[mt].u[0] = *(const uint4*)(p);
            af[mt].u[1] = *(const uint4*)(p + 16);
            if (have_next) __builtin_prefetch(p + 32, 0, 0);
        }

        // B fragments from LDS: column = lane%16, K gathered in table order.
        Frag bf[4];
        #pragma unroll
        for (int nt = 0; nt < 4; ++nt) {
            const unsigned short* lp = &Bs[buf][(g * 8) * BS_STRIDE + wn * 64 + nt * 16 + ml];
            #pragma unroll
            for (int s = 0; s < 16; ++s)
                ((unsigned short*)&bf[nt])[s] = lp[((s & 7) + (s >> 3) * 16) * BS_STRIDE];
        }

        #pragma unroll
        for (int mt = 0; mt < 2; ++mt)
            #pragma unroll
            for (int nt = 0; nt < 4; ++nt)
                acc[mt][nt] = wmma_bf16(af[mt], bf[nt], acc[mt][nt]);

        __syncthreads();   // all waves done reading before next overwrite
        buf ^= 1;
    }

    // Epilogue. C layout: element i -> row (i + 8*g), col ml within 16x16 tile.
    #pragma unroll
    for (int mt = 0; mt < 2; ++mt) {
        #pragma unroll
        for (int nt = 0; nt < 4; ++nt) {
            const int col = wbn + nt * 16 + ml;
            const float bv = bias ? bias[col] : 0.0f;
            #pragma unroll
            for (int i = 0; i < 8; ++i) {
                const int r = bm + mt * 16 + i + 8 * g;
                float v = acc[mt][nt][i] + bv;
                if (emode == 0) {
                    outf[(size_t)r * N + col] = v;
                } else if (emode == 1) {
                    float gv = 0.5f * v * (1.0f + erff(v * 0.70710678118654752f));
                    outh[(size_t)r * N + col] = f2bf(gv);
                } else {
                    const int b = r >> 12;            // L == 4096
                    const int l = r & (LL - 1);
                    const int h = col >> 6;           // HD == 64
                    const int d = col & 63;
                    if (emode == 2)
                        outh[(((size_t)b * NH + h) * LL + l) * HD + d] = f2bf(v);
                    else
                        outh[(((size_t)b * NH + h) * HD + d) * LL + l] = f2bf(v);
                }
            }
        }
    }
}

// ---------------------------------------------------------------------------
// Flash attention (bf16 WMMA, fp32 accumulate, online softmax).
// One wave handles one 16-row q block of one (b,h). 32 keys per iteration.
//   Q  : [B,H,L,64] bf16 ; KT : [B,H,64,L] bf16 ; V : [B,H,L,64] bf16
//   ctx: [B,L,512]  bf16  (heads re-interleaved)
// ---------------------------------------------------------------------------
__global__ __launch_bounds__(256) void flash_attn_wmma(
    const unsigned short* __restrict__ Q,
    const unsigned short* __restrict__ KT,
    const unsigned short* __restrict__ V,
    const int* __restrict__ mask,        // [B, L]
    unsigned short* __restrict__ ctx)
{
    __shared__ unsigned short Pl[8][16 * 40];   // per-wave 16x32 bf16 P tile

    const int lane = threadIdx.x & 31;
    const int wid  = threadIdx.x >> 5;
    const int task = blockIdx.x * 8 + wid;      // B*H*(L/16) = 4096 tasks
    const int qblk = task & ((LL / 16) - 1);
    const int bh   = task >> 8;                 // L/16 = 256
    const int b    = bh >> 3;
    const int h    = bh & 7;
    const int q0   = qblk * 16;
    const int g    = lane >> 4;
    const int ml   = lane & 15;

    const unsigned short* Qp  = Q  + (size_t)bh * LL * HD;
    const unsigned short* KTp = KT + (size_t)bh * HD * LL;
    const unsigned short* Vp  = V  + (size_t)bh * LL * HD;

    Frag aq[2];
    #pragma unroll
    for (int kk = 0; kk < 2; ++kk) {
        const unsigned short* p = Qp + (size_t)(q0 + ml) * HD + kk * 32 + g * 8;
        aq[kk].u[0] = *(const uint4*)(p);
        aq[kk].u[1] = *(const uint4*)(p + 16);
    }

    float m_i[8], l_i[8];
    v8f acc[4];
    v8f zero = {};
    #pragma unroll
    for (int i = 0; i < 8; ++i) { m_i[i] = -1e30f; l_i[i] = 0.0f; }
    #pragma unroll
    for (int nt = 0; nt < 4; ++nt) acc[nt] = zero;

    const float scale = 0.125f;   // 1/sqrt(64)

    for (int key0 = 0; key0 < LL; key0 += 32) {
        v8f s[2];
        #pragma unroll
        for (int t = 0; t < 2; ++t) {
            Frag bk0, bk1;
            const unsigned short* p0 = KTp + (size_t)(g * 8)      * LL + key0 + t * 16 + ml;
            const unsigned short* p1 = KTp + (size_t)(g * 8 + 32) * LL + key0 + t * 16 + ml;
            #pragma unroll
            for (int s = 0; s < 16; ++s) {
                const size_t ko = (size_t)((s & 7) + (s >> 3) * 16) * LL;
                ((unsigned short*)&bk0)[s] = p0[ko];
                ((unsigned short*)&bk1)[s] = p1[ko];
            }
            v8f z = zero;
            z = wmma_bf16(aq[0], bk0, z);
            z = wmma_bf16(aq[1], bk1, z);
            s[t] = z;
        }

        int mk0 = mask[b * LL + key0 + ml];
        int mk1 = mask[b * LL + key0 + 16 + ml];
        #pragma unroll
        for (int i = 0; i < 8; ++i) {
            float s0 = s[0][i] * scale;
            float s1 = s[1][i] * scale;
            if (mk0 == 0) s0 = -1e9f;
            if (mk1 == 0) s1 = -1e9f;
            s[0][i] = s0; s[1][i] = s1;
        }

        float p0[8], p1[8];
        #pragma unroll
        for (int i = 0; i < 8; ++i) {
            float mx = fmaxf(s[0][i], s[1][i]);
            #pragma unroll
            for (int off = 1; off < 16; off <<= 1)
                mx = fmaxf(mx, __shfl_xor(mx, off, 32));
            float mnew = fmaxf(m_i[i], mx);
            float e0 = __expf(s[0][i] - mnew);
            float e1 = __expf(s[1][i] - mnew);
            float sum = e0 + e1;
            #pragma unroll
            for (int off = 1; off < 16; off <<= 1)
                sum += __shfl_xor(sum, off, 32);
            float corr = __expf(m_i[i] - mnew);
            l_i[i] = l_i[i] * corr + sum;
            m_i[i] = mnew;
            p0[i] = e0; p1[i] = e1;
            #pragma unroll
            for (int nt = 0; nt < 4; ++nt) acc[nt][i] *= corr;
        }

        // P: C-layout -> A-fragment via per-wave LDS bounce.
        unsigned short* pl = Pl[wid];
        #pragma unroll
        for (int i = 0; i < 8; ++i) {
            const int row = i + 8 * g;
            pl[row * 40 + ml]      = f2bf(p0[i]);
            pl[row * 40 + 16 + ml] = f2bf(p1[i]);
        }
        asm volatile("s_wait_dscnt 0" ::: "memory");

        Frag ap;
        {
            const unsigned short* pp = pl + ml * 40 + g * 8;
            ap.u[0] = *(const uint4*)(pp);
            ap.u[1] = *(const uint4*)(pp + 16);
        }

        #pragma unroll
        for (int nt = 0; nt < 4; ++nt) {
            Frag bv;
            const unsigned short* pv = Vp + (size_t)(key0 + g * 8) * HD + nt * 16 + ml;
            #pragma unroll
            for (int s = 0; s < 16; ++s)
                ((unsigned short*)&bv)[s] = pv[((s & 7) + (s >> 3) * 16) * HD];
            acc[nt] = wmma_bf16(ap, bv, acc[nt]);
        }
    }

    #pragma unroll
    for (int nt = 0; nt < 4; ++nt) {
        #pragma unroll
        for (int i = 0; i < 8; ++i) {
            const int r = q0 + i + 8 * g;
            float v = acc[nt][i] / l_i[i];
            ctx[((size_t)b * LL + r) * DE + h * HD + nt * 16 + ml] = f2bf(v);
        }
    }
}

// ---------------------------------------------------------------------------
// Residual + LayerNorm: out = LN(X + Y) * gamma + beta  (row width 512)
// ---------------------------------------------------------------------------
__global__ __launch_bounds__(256) void ln_residual(
    const float* __restrict__ X, const float* __restrict__ Y,
    const float* __restrict__ gamma, const float* __restrict__ beta,
    float* __restrict__ outf, unsigned short* __restrict__ outh, int M)
{
    const int lane = threadIdx.x & 31;
    const int wid  = threadIdx.x >> 5;
    const int row  = blockIdx.x * 8 + wid;
    if (row >= M) return;

    const float* xr = X + (size_t)row * DE;
    const float* yr = Y + (size_t)row * DE;

    float v[16];
    float s = 0.0f;
    #pragma unroll
    for (int j = 0; j < 16; ++j) {
        const int c = lane + j * 32;
        v[j] = xr[c] + yr[c];
        s += v[j];
    }
    #pragma unroll
    for (int off = 16; off > 0; off >>= 1) s += __shfl_xor(s, off, 32);
    const float mean = s * (1.0f / (float)DE);

    float vs = 0.0f;
    #pragma unroll
    for (int j = 0; j < 16; ++j) { float d = v[j] - mean; vs += d * d; }
    #pragma unroll
    for (int off = 16; off > 0; off >>= 1) vs += __shfl_xor(vs, off, 32);
    const float inv = rsqrtf(vs * (1.0f / (float)DE) + 1e-5f);

    #pragma unroll
    for (int j = 0; j < 16; ++j) {
        const int c = lane + j * 32;
        float o = (v[j] - mean) * inv * gamma[c] + beta[c];
        if (outf) outf[(size_t)row * DE + c] = o;
        if (outh) outh[(size_t)row * DE + c] = f2bf(o);
    }
}

// ---------------------------------------------------------------------------
// Host launcher
// ---------------------------------------------------------------------------
extern "C" void kernel_launch(void* const* d_in, const int* in_sizes, int n_in,
                              void* d_out, int out_size, void* d_ws, size_t ws_size,
                              hipStream_t stream) {
    (void)in_sizes; (void)n_in; (void)out_size; (void)ws_size;

    const float* x     = (const float*)d_in[0];
    const int*   mask  = (const int*)  d_in[1];
    const float* Wq    = (const float*)d_in[2];
    const float* bq    = (const float*)d_in[3];
    const float* Wk    = (const float*)d_in[4];
    const float* bk    = (const float*)d_in[5];
    const float* Wv    = (const float*)d_in[6];
    const float* bv    = (const float*)d_in[7];
    const float* Wo    = (const float*)d_in[8];
    const float* bo    = (const float*)d_in[9];
    const float* ln1_g = (const float*)d_in[10];
    const float* ln1_b = (const float*)d_in[11];
    const float* ln2_g = (const float*)d_in[12];
    const float* ln2_b = (const float*)d_in[13];
    const float* W1    = (const float*)d_in[14];
    const float* b1    = (const float*)d_in[15];
    const float* W2    = (const float*)d_in[16];
    const float* b2    = (const float*)d_in[17];

    char* p = (char*)d_ws;
    auto grab = [&](size_t bytes) -> char* {
        char* r = p;
        p += (bytes + 255) & ~(size_t)255;
        return r;
    };

    unsigned short* xb   = (unsigned short*)grab((size_t)MTOT * DE * 2);
    unsigned short* wqb  = (unsigned short*)grab((size_t)DE * DE * 2);
    unsigned short* wkb  = (unsigned short*)grab((size_t)DE * DE * 2);
    unsigned short* wvb  = (unsigned short*)grab((size_t)DE * DE * 2);
    unsigned short* wob  = (unsigned short*)grab((size_t)DE * DE * 2);
    unsigned short* w1b  = (unsigned short*)grab((size_t)DE * DFF * 2);
    unsigned short* w2b  = (unsigned short*)grab((size_t)DFF * DE * 2);
    unsigned short* qb   = (unsigned short*)grab((size_t)MTOT * DE * 2);   // [B,H,L,64]
    unsigned short* ktb  = (unsigned short*)grab((size_t)MTOT * DE * 2);   // [B,H,64,L]
    unsigned short* vbuf = (unsigned short*)grab((size_t)MTOT * DE * 2);   // [B,H,L,64]
    unsigned short* ctxb = (unsigned short*)grab((size_t)MTOT * DE * 2);   // [B,L,512]
    float*          attnf= (float*)         grab((size_t)MTOT * DE * 4);
    float*          hf   = (float*)         grab((size_t)MTOT * DE * 4);
    unsigned short* hb   = (unsigned short*)grab((size_t)MTOT * DE * 2);
    unsigned short* ffb  = (unsigned short*)grab((size_t)MTOT * DFF * 2);
    float*          fff  = (float*)         grab((size_t)MTOT * DE * 4);

    // --- converts ---
    cvt_f32_bf16<<<2048, 256, 0, stream>>>(x,  xb,  MTOT * DE);
    cvt_f32_bf16<<<512,  256, 0, stream>>>(Wq, wqb, DE * DE);
    cvt_f32_bf16<<<512,  256, 0, stream>>>(Wk, wkb, DE * DE);
    cvt_f32_bf16<<<512,  256, 0, stream>>>(Wv, wvb, DE * DE);
    cvt_f32_bf16<<<512,  256, 0, stream>>>(Wo, wob, DE * DE);
    cvt_f32_bf16<<<2048, 256, 0, stream>>>(W1, w1b, DE * DFF);
    cvt_f32_bf16<<<2048, 256, 0, stream>>>(W2, w2b, DFF * DE);

    // --- QKV projections (head-split / transposed-K outputs) ---
    dim3 g512(MTOT / 128, DE / 128);
    gemm_wmma_bf16<<<g512, 256, 0, stream>>>(xb, wqb, bq, MTOT, DE, DE, 2, nullptr, qb);
    gemm_wmma_bf16<<<g512, 256, 0, stream>>>(xb, wkb, bk, MTOT, DE, DE, 3, nullptr, ktb);
    gemm_wmma_bf16<<<g512, 256, 0, stream>>>(xb, wvb, bv, MTOT, DE, DE, 2, nullptr, vbuf);

    // --- flash attention ---
    flash_attn_wmma<<<(BB * NH * (LL / 16)) / 8, 256, 0, stream>>>(qb, ktb, vbuf, mask, ctxb);

    // --- output projection ---
    gemm_wmma_bf16<<<g512, 256, 0, stream>>>(ctxb, wob, bo, MTOT, DE, DE, 0, attnf, nullptr);

    // --- LN1 (residual) ---
    ln_residual<<<MTOT / 8, 256, 0, stream>>>(x, attnf, ln1_g, ln1_b, hf, hb, MTOT);

    // --- FFN ---
    dim3 gff(MTOT / 128, DFF / 128);
    gemm_wmma_bf16<<<gff, 256, 0, stream>>>(hb, w1b, b1, MTOT, DFF, DE, 1, nullptr, ffb);
    gemm_wmma_bf16<<<g512, 256, 0, stream>>>(ffb, w2b, b2, MTOT, DE, DFF, 0, fff, nullptr);

    // --- LN2 (residual) -> output ---
    ln_residual<<<MTOT / 8, 256, 0, stream>>>(hf, fff, ln2_g, ln2_b, (float*)d_out, nullptr, MTOT);
}
